// distance_ll_59227599011949
// MI455X (gfx1250) — matrix operand order repaired and checked by reading, not compiled, not run on hardware
//
#include <hip/hip_runtime.h>
#include <hip/hip_bf16.h>

// ---------------- problem constants ----------------
#define BATCH   8
#define N_L     64
#define N_R     512
#define F_DIM   128
#define HID     128
#define KMIX    10
#define P_PAIRS (BATCH * N_L * N_R)      // 262144
#define M_MDN   131072
#define CONTACT_TH 8.0f
#define BN_EPS  1e-5f

// output offsets (floats) in d_out (concat in return order)
#define OFF_LOGPI   0
#define OFF_SIGMA   ((size_t)M_MDN * KMIX)                 // 1310720
#define OFF_MU      ((size_t)2 * M_MDN * KMIX)             // 2621440
#define OFF_PROB    ((size_t)3 * M_MDN * KMIX)             // 3932160
#define OFF_CONTACT (OFF_PROB + BATCH)                     // 3932168
#define OFF_TRUTH   (OFF_CONTACT + P_PAIRS)                // 4194312

#define STAGE3_BLOCKS (P_PAIRS / 64)  // 4096 (4 waves x 16 rows per block)
#define MDN_BLOCKS    (M_MDN / 64)    // 2048

typedef __attribute__((ext_vector_type(16))) _Float16 v16h;
typedef __attribute__((ext_vector_type(8)))  float    v8f;

union FragU { v16h h; unsigned u[8]; };

// ---- WMMA wrapper (codegen-confirmed builtin) ----
__device__ __forceinline__ v8f wmma_f16(v16h a, v16h b, v8f c) {
  return __builtin_amdgcn_wmma_f32_16x16x32_f16(false, a, false, b, (short)0, c, false, false);
}

// A fragment (16-bit 16x32), row-major f16 tile (pitch halves).
// lanes 0-15: row=lane, K = k0+{0..7,16..23}; lanes 16-31: row=lane-16, K = k0+{8..15,24..31}
__device__ __forceinline__ v16h load_a_frag(const _Float16* base, int pitch, int row0, int k0, int lane) {
  int l15 = lane & 15;
  int kb  = k0 + ((lane & 16) ? 8 : 0);
  const _Float16* p = base + (size_t)(row0 + l15) * pitch + kb;
  FragU f;
#pragma unroll
  for (int r = 0; r < 4; ++r) f.u[r]     = *(const unsigned*)(p + 2 * r);
#pragma unroll
  for (int r = 0; r < 4; ++r) f.u[4 + r] = *(const unsigned*)(p + 16 + 2 * r);
  return f.h;
}

// B fragment (16-bit 32x16) from pair-interleaved weights: u32 at [k/2]*npitch + n holds (K=k,K=k+1).
// lane n: VGPR r -> K = k0 + 8*(lane>>4) + 2r (r<4), k0 + 16 + 8*(lane>>4) + 2(r-4) (r>=4)
__device__ __forceinline__ v16h load_b_frag(const unsigned* bint, int npitch, int k0, int n0, int lane) {
  int n  = n0 + (lane & 15);
  int kp = (k0 >> 1) + ((lane & 16) ? 4 : 0);
  FragU f;
#pragma unroll
  for (int r = 0; r < 8; ++r) f.u[r] = bint[(size_t)(kp + r + (r & 4)) * npitch + n];
  return f.h;
}

__device__ __forceinline__ unsigned pack2(float a, float b) {
  union { _Float16 h[2]; unsigned u; } x;
  x.h[0] = (_Float16)a; x.h[1] = (_Float16)b; return x.u;
}
__device__ __forceinline__ float eluf(float x) { return x > 0.f ? x : (__expf(x) - 1.f); }

// ---------------- prep kernels ----------------
__global__ __launch_bounds__(256) void cvt_f16_kernel(const float* src, _Float16* dst, int n) {
  int i = blockIdx.x * 256 + threadIdx.x;
  if (i < n) dst[i] = (_Float16)src[i];
}

__global__ __launch_bounds__(256) void prep_weights_kernel(
    const float* W1, const float* Wc1, const float* Wpi, const float* Wsig, const float* Wmu,
    unsigned* w1top, unsigned* w1bot, unsigned* wc1i, unsigned* walli) {
  int idx = blockIdx.x * 256 + threadIdx.x;
  if (idx < 64 * 128) {
    int kp = idx / 128, n = idx % 128;
    w1top[idx] = pack2(W1[(2 * kp) * 128 + n],       W1[(2 * kp + 1) * 128 + n]);
    w1bot[idx] = pack2(W1[(128 + 2 * kp) * 128 + n], W1[(129 + 2 * kp) * 128 + n]);
    wc1i[idx]  = pack2(Wc1[(2 * kp) * 128 + n],      Wc1[(2 * kp + 1) * 128 + n]);
  }
  if (idx < 64 * 48) {
    int kp = idx / 48, col = idx % 48;
    int head = col >> 4, c = col & 15;
    const float* W = head == 0 ? Wpi : (head == 1 ? Wsig : Wmu);
    float a = (c < KMIX) ? W[(2 * kp) * KMIX + c]     : 0.f;
    float b = (c < KMIX) ? W[(2 * kp + 1) * KMIX + c] : 0.f;
    walli[idx] = pack2(a, b);
  }
}

// ---------------- stage 1: Lpart / Rpart small WMMA GEMMs (M x 128 x 128) ----------------
__global__ __launch_bounds__(128) void gemm_part_kernel(const _Float16* A, const unsigned* Bint, float* C) {
  __shared__ unsigned bl[64 * 128];
  int tid = threadIdx.x, wave = tid >> 5, lane = tid & 31;
  for (int i = tid; i < 64 * 128; i += 128) bl[i] = Bint[i];
  __syncthreads();
  int row0 = blockIdx.x * 64 + wave * 16;
  v8f acc[8];
#pragma unroll
  for (int t = 0; t < 8; ++t)
#pragma unroll
    for (int v = 0; v < 8; ++v) acc[t][v] = 0.f;
#pragma unroll
  for (int ks = 0; ks < 4; ++ks) {
    v16h a = load_a_frag(A, 128, row0, ks * 32, lane);
#pragma unroll
    for (int t = 0; t < 8; ++t) {
      v16h b = load_b_frag(bl, 128, ks * 32, t * 16, lane);
      acc[t] = wmma_f16(a, b, acc[t]);
    }
  }
  int l15 = lane & 15, rhi = (lane & 16) ? 8 : 0;
#pragma unroll
  for (int t = 0; t < 8; ++t)
#pragma unroll
    for (int v = 0; v < 8; ++v)
      C[(size_t)(row0 + rhi + v) * 128 + t * 16 + l15] = acc[t][v];
}

// ---------------- stage 2: BN1 closed-form stats ----------------
__global__ __launch_bounds__(128) void col_stats_kernel(const float* X, float* S, float* S2, int rowsPerB) {
  int b = blockIdx.x, c = threadIdx.x;
  const float* p = X + (size_t)b * rowsPerB * 128 + c;
  float s = 0.f, s2 = 0.f;
  for (int r = 0; r < rowsPerB; ++r) { float v = p[(size_t)r * 128]; s += v; s2 += v * v; }
  S[b * 128 + c] = s; S2[b * 128 + c] = s2;
}

__global__ __launch_bounds__(128) void bn1_fin_kernel(
    const float* SL, const float* SL2, const float* SR, const float* SR2,
    const float* b1, const float* g1, const float* beta1, float* bn1c) {
  int c = threadIdx.x;
  float sLR = 0.f, sLR2 = 0.f;
  for (int b = 0; b < BATCH; ++b) {
    float sl = SL[b * 128 + c], sr = SR[b * 128 + c];
    sLR  += (float)N_R * sl + (float)N_L * sr;
    sLR2 += (float)N_R * SL2[b * 128 + c] + (float)N_L * SR2[b * 128 + c] + 2.f * sl * sr;
  }
  float Pf = (float)P_PAIRS, bb = b1[c];
  float sumz  = sLR + Pf * bb;
  float sumz2 = sLR2 + 2.f * bb * sLR + Pf * bb * bb;
  float mu  = sumz / Pf;
  float var = sumz2 / Pf - mu * mu;
  float a   = g1[c] * rsqrtf(var + BN_EPS);
  bn1c[c]       = a;                         // scale
  bn1c[128 + c] = a * (bb - mu) + beta1[c];  // offset
}

// ---------------- pair distances + contact truth ----------------
__global__ __launch_bounds__(256) void pairdist_kernel(const float* pl, const float* pr,
                                                       float* dist, float* truth) {
  int p = blockIdx.x * 256 + threadIdx.x;
  int b = p >> 15, i = (p >> 9) & 63, j = p & 511;
  const float* a = pl + (size_t)(b * N_L + i) * 3;
  const float* q = pr + (size_t)(b * N_R + j) * 3;
  float dx = a[0] - q[0], dy = a[1] - q[1], dz = a[2] - q[2];
  float d = sqrtf(dx * dx + dy * dy + dz * dz);
  dist[p]  = d;
  truth[p] = (d < CONTACT_TH) ? 1.0f : 0.0f;
}

// ---------------- stage 3: fused h(on-the-fly) @ Wc1 via WMMA, y_f16 + BN2 partials ----------------
__global__ __launch_bounds__(128) void stage3_kernel(
    const float* Lpart, const float* Rpart, const float* bn1c,
    const unsigned* wc1i, const float* bc1, _Float16* y16, float* bn2p) {
  __shared__ unsigned bl[64 * 128];
  __shared__ __align__(16) _Float16 ht[4][16 * 128];
  __shared__ float wsum[4][2][128];
  int tid = threadIdx.x, wave = tid >> 5, lane = tid & 31;
  for (int i = tid; i < 64 * 128; i += 128) bl[i] = wc1i[i];

  int p0 = blockIdx.x * 64 + wave * 16;                // 16 pairs share (b,i), j = j0..j0+15
  int b = p0 >> 15, i = (p0 >> 9) & 63, j0 = p0 & 511;
  const float* Lrow  = Lpart + (size_t)(b * N_L + i) * 128;
  const float* Rbase = Rpart + (size_t)(b * N_R + j0) * 128;
#pragma unroll 1
  for (int m = 0; m < 16; ++m) {
    const float* R = Rbase + (size_t)m * 128;
#pragma unroll
    for (int t = 0; t < 4; ++t) {
      int c = lane + 32 * t;
      float z = bn1c[c] * (Lrow[c] + R[c]) + bn1c[128 + c];
      ht[wave][m * 128 + c] = (_Float16)eluf(z);
    }
  }
  __syncthreads();

  int l15 = lane & 15, rhi = (lane & 16) ? 8 : 0;
  v8f acc[8];
#pragma unroll
  for (int t = 0; t < 8; ++t) {
    float bv = bc1[t * 16 + l15];
#pragma unroll
    for (int v = 0; v < 8; ++v) acc[t][v] = bv;
  }
#pragma unroll
  for (int ks = 0; ks < 4; ++ks) {
    v16h a = load_a_frag(&ht[wave][0], 128, 0, ks * 32, lane);
#pragma unroll
    for (int t = 0; t < 8; ++t) {
      v16h bf = load_b_frag(bl, 128, ks * 32, t * 16, lane);
      acc[t] = wmma_f16(a, bf, acc[t]);
    }
  }
#pragma unroll
  for (int t = 0; t < 8; ++t) {
    float s1 = 0.f, s2 = 0.f;
#pragma unroll
    for (int v = 0; v < 8; ++v) {
      float y = acc[t][v];
      y16[(size_t)(p0 + rhi + v) * 128 + t * 16 + l15] = (_Float16)y;
      s1 += y; s2 += y * y;
    }
    s1 += __shfl_xor(s1, 16, 32);
    s2 += __shfl_xor(s2, 16, 32);
    if (lane < 16) { wsum[wave][0][t * 16 + l15] = s1; wsum[wave][1][t * 16 + l15] = s2; }
  }
  __syncthreads();
  if (tid < 128) {  // deterministic per-block partials
    float s = wsum[0][0][tid] + wsum[1][0][tid] + wsum[2][0][tid] + wsum[3][0][tid];
    float q = wsum[0][1][tid] + wsum[1][1][tid] + wsum[2][1][tid] + wsum[3][1][tid];
    bn2p[(size_t)blockIdx.x * 256 + tid]       = s;
    bn2p[(size_t)blockIdx.x * 256 + 128 + tid] = q;
  }
}

__global__ __launch_bounds__(256) void bn2_fin_kernel(const float* bn2p, const float* gc,
                                                      const float* betac, float* bn2c) {
  int tid = threadIdx.x;
  float s = 0.f;
  for (int blk = 0; blk < STAGE3_BLOCKS; ++blk) s += bn2p[(size_t)blk * 256 + tid];
  __shared__ float red[256];
  red[tid] = s;
  __syncthreads();
  if (tid < 128) {
    float Pf = (float)P_PAIRS;
    float mu = red[tid] / Pf;
    float var = red[128 + tid] / Pf - mu * mu;
    float a = gc[tid] * rsqrtf(var + BN_EPS);
    bn2c[tid]       = a;
    bn2c[128 + tid] = betac[tid] - a * mu;
  }
}

// ---------------- contact head ----------------
__global__ __launch_bounds__(256) void contact_kernel(const _Float16* y16, const float* bn2c,
                                                      const float* Wc2, const float* bc2, float* out) {
  int p = blockIdx.x * 256 + threadIdx.x;
  const _Float16* row = y16 + (size_t)p * 128;
  float acc = 0.f;
#pragma unroll 4
  for (int c = 0; c < 128; ++c) {
    float v = bn2c[c] * (float)row[c] + bn2c[128 + c];
    acc += fmaxf(v, 0.f) * Wc2[c];
  }
  out[p] = acc + bc2[0];
}

// ---------------- MDN: gathered-h WMMA + heads + deterministic segment partials ----------------
__global__ __launch_bounds__(128) void mdn_kernel(
    const int* mask, const float* Lpart, const float* Rpart, const float* bn1c,
    const unsigned* walli, const float* bpi, const float* bsig, const float* bmu,
    const float* dist, float* out, float* msum, float* mcnt) {
  __shared__ unsigned bl[64 * 48];
  __shared__ __align__(16) _Float16 ht[4][16 * 128];
  __shared__ float dt[4][16 * 48];
  __shared__ float rowval[64];
  __shared__ int   rowb[64];
  int tid = threadIdx.x, wave = tid >> 5, lane = tid & 31;
  for (int i = tid; i < 64 * 48; i += 128) bl[i] = walli[i];

  int m0 = blockIdx.x * 64 + wave * 16;
#pragma unroll 1
  for (int m = 0; m < 16; ++m) {
    int p = mask[m0 + m];
    int b = p >> 15, i = (p >> 9) & 63, j = p & 511;
    const float* L = Lpart + (size_t)(b * N_L + i) * 128;
    const float* R = Rpart + (size_t)(b * N_R + j) * 128;
#pragma unroll
    for (int t = 0; t < 4; ++t) {
      int c = lane + 32 * t;
      float z = bn1c[c] * (L[c] + R[c]) + bn1c[128 + c];
      ht[wave][m * 128 + c] = (_Float16)eluf(z);
    }
  }
  __syncthreads();

  int l15 = lane & 15, rhi = (lane & 16) ? 8 : 0;
  v8f acc[3];
#pragma unroll
  for (int t = 0; t < 3; ++t) {
    const float* bias = (t == 0) ? bpi : ((t == 1) ? bsig : bmu);
    float bv = (l15 < KMIX) ? bias[l15] : 0.f;
#pragma unroll
    for (int v = 0; v < 8; ++v) acc[t][v] = bv;
  }
#pragma unroll
  for (int ks = 0; ks < 4; ++ks) {
    v16h a = load_a_frag(&ht[wave][0], 128, 0, ks * 32, lane);
#pragma unroll
    for (int t = 0; t < 3; ++t) {
      v16h bf = load_b_frag(bl, 48, ks * 32, t * 16, lane);
      acc[t] = wmma_f16(a, bf, acc[t]);
    }
  }
#pragma unroll
  for (int t = 0; t < 3; ++t)
#pragma unroll
    for (int v = 0; v < 8; ++v)
      dt[wave][(rhi + v) * 48 + t * 16 + l15] = acc[t][v];
  __syncthreads();

  if (lane < 16) {
    int lm = lane;
    int m = m0 + lm;
    int p = mask[m];
    float z[KMIX], mx = -1e30f;
#pragma unroll
    for (int k = 0; k < KMIX; ++k) { z[k] = dt[wave][lm * 48 + k]; mx = fmaxf(mx, z[k]); }
    float se = 0.f;
#pragma unroll
    for (int k = 0; k < KMIX; ++k) se += __expf(z[k] - mx);
    float lse = mx + __logf(se);
    float y = dist[p];
    float amax = -1e30f, terms[KMIX];
#pragma unroll
    for (int k = 0; k < KMIX; ++k) {
      float lp = z[k] - lse;
      out[OFF_LOGPI + (size_t)m * KMIX + k] = lp;
      float sg = eluf(dt[wave][lm * 48 + 16 + k]) + 1.1f;
      float mk = eluf(dt[wave][lm * 48 + 32 + k]) + 1.0f;
      out[OFF_SIGMA + (size_t)m * KMIX + k] = sg;
      out[OFF_MU + (size_t)m * KMIX + k]    = mk;
      float r = (y - mk) / sg;
      float ll = -0.5f * r * r - __logf(sg) - 0.91893853320467274f;  // 0.5*log(2*pi)
      terms[k] = lp + ll;
      amax = fmaxf(amax, terms[k]);
    }
    float s2 = 0.f;
#pragma unroll
    for (int k = 0; k < KMIX; ++k) s2 += __expf(terms[k] - amax);
    rowval[wave * 16 + lm] = -(amax + __logf(s2));
    rowb[wave * 16 + lm] = p >> 15;
  }
  __syncthreads();
  if (tid < BATCH) {  // deterministic fixed-order segment accumulation
    float s = 0.f; int c = 0;
    for (int r = 0; r < 64; ++r)
      if (rowb[r] == tid) { s += rowval[r]; ++c; }
    msum[(size_t)blockIdx.x * BATCH + tid] = s;
    mcnt[(size_t)blockIdx.x * BATCH + tid] = (float)c;
  }
}

__global__ __launch_bounds__(32) void prob_fin_kernel(const float* msum, const float* mcnt, float* out) {
  int b = threadIdx.x;
  if (b >= BATCH) return;
  float s = 0.f, c = 0.f;
  for (int blk = 0; blk < MDN_BLOCKS; ++blk) { s += msum[blk * BATCH + b]; c += mcnt[blk * BATCH + b]; }
  out[OFF_PROB + b] = s / fmaxf(c, 1.f);
}

// ---------------- host-side orchestration ----------------
extern "C" void kernel_launch(void* const* d_in, const int* in_sizes, int n_in,
                              void* d_out, int out_size, void* d_ws, size_t ws_size,
                              hipStream_t stream) {
  (void)in_sizes; (void)n_in; (void)out_size; (void)ws_size;
  const float* feat_lig = (const float*)d_in[0];
  const float* pos_lig  = (const float*)d_in[1];
  const float* feat_rec = (const float*)d_in[3];
  const float* pos_rec  = (const float*)d_in[4];
  const int*   mdn_mask = (const int*)d_in[6];
  const float* W1 = (const float*)d_in[7];
  const float* b1 = (const float*)d_in[8];
  const float* g1 = (const float*)d_in[9];
  const float* beta1 = (const float*)d_in[10];
  const float* Wpi = (const float*)d_in[11];  const float* bpi = (const float*)d_in[12];
  const float* Wsig = (const float*)d_in[13]; const float* bsig = (const float*)d_in[14];
  const float* Wmu = (const float*)d_in[15];  const float* bmu = (const float*)d_in[16];
  const float* Wc1 = (const float*)d_in[17];  const float* bc1 = (const float*)d_in[18];
  const float* gc = (const float*)d_in[19];   const float* betac = (const float*)d_in[20];
  const float* Wc2 = (const float*)d_in[21];  const float* bc2 = (const float*)d_in[22];
  float* out = (float*)d_out;

  // workspace carve-up (256B aligned)
  char* w = (char*)d_ws;
  size_t o = 0;
  auto carve = [&](size_t bytes) { char* r = w + o; o = (o + bytes + 255) & ~(size_t)255; return r; };
  _Float16* featL16 = (_Float16*)carve((size_t)BATCH * N_L * F_DIM * 2);
  _Float16* featR16 = (_Float16*)carve((size_t)BATCH * N_R * F_DIM * 2);
  unsigned* w1top   = (unsigned*)carve(64 * 128 * 4);
  unsigned* w1bot   = (unsigned*)carve(64 * 128 * 4);
  unsigned* wc1i    = (unsigned*)carve(64 * 128 * 4);
  unsigned* walli   = (unsigned*)carve(64 * 48 * 4);
  float* Lpart = (float*)carve((size_t)BATCH * N_L * HID * 4);
  float* Rpart = (float*)carve((size_t)BATCH * N_R * HID * 4);
  float* SL  = (float*)carve(BATCH * 128 * 4);
  float* SL2 = (float*)carve(BATCH * 128 * 4);
  float* SR  = (float*)carve(BATCH * 128 * 4);
  float* SR2 = (float*)carve(BATCH * 128 * 4);
  float* bn1c = (float*)carve(256 * 4);
  float* dist = (float*)carve((size_t)P_PAIRS * 4);
  _Float16* y16 = (_Float16*)carve((size_t)P_PAIRS * HID * 2);
  float* bn2p = (float*)carve((size_t)STAGE3_BLOCKS * 256 * 4);
  float* bn2c = (float*)carve(256 * 4);
  float* msum = (float*)carve((size_t)MDN_BLOCKS * BATCH * 4);
  float* mcnt = (float*)carve((size_t)MDN_BLOCKS * BATCH * 4);

  // prep: f16 conversion + pair-interleaved weights
  cvt_f16_kernel<<<(BATCH * N_L * F_DIM + 255) / 256, 256, 0, stream>>>(feat_lig, featL16, BATCH * N_L * F_DIM);
  cvt_f16_kernel<<<(BATCH * N_R * F_DIM + 255) / 256, 256, 0, stream>>>(feat_rec, featR16, BATCH * N_R * F_DIM);
  prep_weights_kernel<<<(64 * 128 + 255) / 256, 256, 0, stream>>>(W1, Wc1, Wpi, Wsig, Wmu, w1top, w1bot, wc1i, walli);

  // stage 1: rank-structured factor GEMMs (WMMA)
  gemm_part_kernel<<<(BATCH * N_L) / 64, 128, 0, stream>>>(featL16, w1top, Lpart);
  gemm_part_kernel<<<(BATCH * N_R) / 64, 128, 0, stream>>>(featR16, w1bot, Rpart);

  // stage 2: closed-form BN1 stats
  col_stats_kernel<<<BATCH, 128, 0, stream>>>(Lpart, SL, SL2, N_L);
  col_stats_kernel<<<BATCH, 128, 0, stream>>>(Rpart, SR, SR2, N_R);
  bn1_fin_kernel<<<1, 128, 0, stream>>>(SL, SL2, SR, SR2, b1, g1, beta1, bn1c);

  // distances + contact truth
  pairdist_kernel<<<P_PAIRS / 256, 256, 0, stream>>>(pos_lig, pos_rec, dist, out + OFF_TRUTH);

  // stage 3: fused h -> y (WMMA), y_f16 + BN2 partials
  stage3_kernel<<<STAGE3_BLOCKS, 128, 0, stream>>>(Lpart, Rpart, bn1c, wc1i, bc1, y16, bn2p);
  bn2_fin_kernel<<<1, 256, 0, stream>>>(bn2p, gc, betac, bn2c);

  // contact head
  contact_kernel<<<P_PAIRS / 256, 256, 0, stream>>>(y16, bn2c, Wc2, bc2, out + OFF_CONTACT);

  // MDN (WMMA heads) + deterministic segment mean
  mdn_kernel<<<MDN_BLOCKS, 128, 0, stream>>>(mdn_mask, Lpart, Rpart, bn1c, walli,
                                             bpi, bsig, bmu, dist, out, msum, mcnt);
  prob_fin_kernel<<<1, 32, 0, stream>>>(msum, mcnt, out);
}
